// Model_81664508166959
// MI455X (gfx1250) — matrix operand order repaired and checked by reading, not compiled, not run on hardware
//
#include <hip/hip_runtime.h>
#include <math.h>

// ---- problem constants (match reference) ----
#define BB    256
#define CC    3
#define IMSZv 224
#define GZv   32
#define TT    7
#define NFv   256
#define NHv   512
#define CLSv  100
#define K1    3072          // C*GZ*GZ
#define KCH   512           // glimpse K chunk held in LDS
#define NCHUNK (K1 / KCH)   // 6

// bf16 weight-cache layout in workspace (element offsets; all 32B-aligned)
#define OFS_WG  0
#define OFS_WF  (OFS_WG  + NFv * K1)    // 786432
#define OFS_WXH (OFS_WF  + NFv * NFv)   // 851968
#define OFS_WHH (OFS_WXH + NHv * NFv)   // 983040
#define OFS_WC  (OFS_WHH + NHv * NHv)   // 1245184
#define W_TOTAL (OFS_WC  + CLSv * NHv)  // 1296384 elements

typedef __attribute__((ext_vector_type(16))) __bf16 v16bf;
typedef __attribute__((ext_vector_type(8)))  float  v8f;
typedef __attribute__((ext_vector_type(4)))  float  v4f;

__device__ __forceinline__ v8f zero8() {
  v8f z = {0.f, 0.f, 0.f, 0.f, 0.f, 0.f, 0.f, 0.f};
  return z;
}

// A fragment (16x32 bf16, contiguous k order): lane holds row = lane&15,
// K = kbase + hiA + {0..15}, hiA = (lane>=16)?16:0.
// p0 must already point at  row*ldk + kbase + hiA.
__device__ __forceinline__ v16bf ldsFragA(const __bf16* p0) {
  v16bf a;
#pragma unroll
  for (int e = 0; e < 16; ++e) { a[e] = p0[e]; }
  return a;
}

// B fragment (32x16 bf16): lane holds column n = lane&15,
// K = kbase + ((lane>=16)?16:0) + {0..15}  (contiguous run, same order as A).
// p0 must point at  n*ldk + kbase + hiB  in a row-major [N][K] weight.
__device__ __forceinline__ v16bf glbFragB(const float* p0) {   // f32 weights: convert
  v16bf b;
#pragma unroll
  for (int e = 0; e < 16; ++e) { b[e] = (__bf16)p0[e]; }
  return b;
}
__device__ __forceinline__ v16bf glbFragB(const __bf16* p0) {  // cached bf16: direct 32B load
  return *(const v16bf*)p0;
}

__device__ __forceinline__ v16bf zeroFragB() {
  v16bf b;
#pragma unroll
  for (int e = 0; e < 16; ++e) { b[e] = (__bf16)0.f; }
  return b;
}

#define WMMA_BF16(C_, A_, B_) \
  __builtin_amdgcn_wmma_f32_16x16x32_bf16(false, (A_), false, (B_), (short)0, (C_), false, false)

__global__ void zero_out_k(float* out) {
  if (threadIdx.x < 8) out[threadIdx.x] = 0.f;
}

// f32 -> bf16 weight conversion (n is a multiple of 4)
__global__ void cvt_bf16_k(const float* __restrict__ src, __bf16* __restrict__ dst, int n4) {
  int i = blockIdx.x * blockDim.x + threadIdx.x;
  if (i < n4) {
    v4f v = ((const v4f*)src)[i];
    __bf16* d = dst + i * 4;
    d[0] = (__bf16)v.x; d[1] = (__bf16)v.y; d[2] = (__bf16)v.z; d[3] = (__bf16)v.w;
  }
}

template <typename WT>
__global__ __launch_bounds__(256)
void ram_fused(const float* __restrict__ x, const int* __restrict__ y,
               const int* __restrict__ locs,
               const WT* __restrict__ Wg,  const float* __restrict__ bg,
               const float* __restrict__ Wl, const float* __restrict__ bl,
               const WT* __restrict__ Wf,  const float* __restrict__ bf,
               const WT* __restrict__ Wxh, const WT* __restrict__ Whh,
               const float* __restrict__ brnn,
               const WT* __restrict__ Wc,  const float* __restrict__ bc,
               float* __restrict__ out) {
  __shared__ __bf16 Ag[16][KCH];       // glimpse K-chunk          16 KB
  __shared__ __bf16 mA[16][NFv];       // relu(g)*relu(lf)          8 KB
  __shared__ __bf16 featA[16][NFv];    // fused feature             8 KB
  __shared__ __bf16 hidA[16][NHv];     // recurrent state          16 KB
  __shared__ float  logitsS[16][112];  // padded logits             7 KB
  __shared__ int    lrI[16][2];
  __shared__ float  lrF[16][2];

  const int tid  = threadIdx.x;
  const int lane = tid & 31;
  const int w    = tid >> 5;           // wave id 0..7
  const int m0   = blockIdx.x * 16;    // batch-row block
  const int nrow = lane & 15;          // row (A) or column (B/D) within tile
  const int hiA  = (lane >= 16) ? 16 : 0;  // A k-offset
  const int hiB  = (lane >= 16) ? 16 : 0;  // B k-offset
  const int hiD  = (lane >= 16) ? 8  : 0;  // D m-offset

  // hidden = 0
  for (int i = tid; i < 16 * NHv; i += 256) (&hidA[0][0])[i] = (__bf16)0.f;

  float lossAcc = 0.f;

  for (int t = 0; t < TT; ++t) {
    // per-step locations for this row block
    if (tid < 32) {
      int r = tid >> 1, comp = tid & 1;
      int v = locs[((t * BB) + m0 + r) * 2 + comp];
      lrI[r][comp] = v;
      lrF[r][comp] = (float)v;
    }

    // ---------- GEMM1: g = glimpse @ Wg^T (K=3072, chunked), fused lf ----------
    v8f acc0 = zero8(), acc1 = zero8();
    for (int kc = 0; kc < NCHUNK; ++kc) {
      __syncthreads();  // prev chunk consumed (also publishes lrI/lrF)
      for (int i = tid; i < 16 * KCH; i += 256) {
        int r  = i >> 9;            // /KCH
        int kk = i & (KCH - 1);
        int k  = kc * KCH + kk;
        int c  = k >> 10;           // /1024
        int rem = k & 1023;
        int gy = rem >> 5, gx = rem & 31;
        size_t src = ((size_t)((m0 + r) * CC + c) * IMSZv + (lrI[r][0] + gy)) * IMSZv
                     + (size_t)(lrI[r][1] + gx);
        Ag[r][kk] = (__bf16)x[src];
      }
      __syncthreads();
#pragma unroll 4
      for (int ks = 0; ks < KCH / 32; ++ks) {
        v16bf a = ldsFragA(&Ag[nrow][ks * 32 + hiA]);
        int kg = kc * KCH + ks * 32;
        v16bf b0 = glbFragB(Wg + (size_t)(w * 16 + nrow) * K1 + kg + hiB);
        acc0 = WMMA_BF16(acc0, a, b0);
        v16bf b1 = glbFragB(Wg + (size_t)((w + 8) * 16 + nrow) * K1 + kg + hiB);
        acc1 = WMMA_BF16(acc1, a, b1);
      }
    }
    // mA = relu(g + bg) * relu(loc @ Wl^T + bl)
#pragma unroll
    for (int ti = 0; ti < 2; ++ti) {
      int n = (w + 8 * ti) * 16 + nrow;
      float bgn = bg[n], wl0 = Wl[2 * n], wl1 = Wl[2 * n + 1], blv = bl[n];
      v8f acc = ti ? acc1 : acc0;
#pragma unroll
      for (int r = 0; r < 8; ++r) {
        int m = r + hiD;
        float g  = acc[r] + bgn;                            g  = g  > 0.f ? g  : 0.f;
        float lf = lrF[m][0] * wl0 + lrF[m][1] * wl1 + blv; lf = lf > 0.f ? lf : 0.f;
        mA[m][n] = (__bf16)(g * lf);
      }
    }
    __syncthreads();

    // ---------- GEMM2: feat = relu(mA @ Wf^T + bf) ----------
    v8f f0 = zero8(), f1 = zero8();
#pragma unroll
    for (int ks = 0; ks < NFv / 32; ++ks) {
      v16bf a = ldsFragA(&mA[nrow][ks * 32 + hiA]);
      v16bf b0 = glbFragB(Wf + (size_t)(w * 16 + nrow) * NFv + ks * 32 + hiB);
      f0 = WMMA_BF16(f0, a, b0);
      v16bf b1 = glbFragB(Wf + (size_t)((w + 8) * 16 + nrow) * NFv + ks * 32 + hiB);
      f1 = WMMA_BF16(f1, a, b1);
    }
#pragma unroll
    for (int ti = 0; ti < 2; ++ti) {
      int n = (w + 8 * ti) * 16 + nrow;
      float bfv = bf[n];
      v8f acc = ti ? f1 : f0;
#pragma unroll
      for (int r = 0; r < 8; ++r) {
        float v = acc[r] + bfv;
        featA[r + hiD][n] = (__bf16)(v > 0.f ? v : 0.f);
      }
    }
    __syncthreads();

    // ---------- GEMM3: hidden = relu(feat @ Wxh^T + hidden @ Whh^T + brnn) ----------
    v8f h[4];
    h[0] = h[1] = h[2] = h[3] = zero8();
#pragma unroll
    for (int ks = 0; ks < NFv / 32; ++ks) {
      v16bf a = ldsFragA(&featA[nrow][ks * 32 + hiA]);
#pragma unroll
      for (int j = 0; j < 4; ++j) {
        int n = (w * 4 + j) * 16 + nrow;
        v16bf b = glbFragB(Wxh + (size_t)n * NFv + ks * 32 + hiB);
        h[j] = WMMA_BF16(h[j], a, b);
      }
    }
#pragma unroll
    for (int ks = 0; ks < NHv / 32; ++ks) {
      v16bf a = ldsFragA(&hidA[nrow][ks * 32 + hiA]);
#pragma unroll
      for (int j = 0; j < 4; ++j) {
        int n = (w * 4 + j) * 16 + nrow;
        v16bf b = glbFragB(Whh + (size_t)n * NHv + ks * 32 + hiB);
        h[j] = WMMA_BF16(h[j], a, b);
      }
    }
    __syncthreads();  // all reads of hidA complete before overwrite
#pragma unroll
    for (int j = 0; j < 4; ++j) {
      int n = (w * 4 + j) * 16 + nrow;
      float bv = brnn[n];
#pragma unroll
      for (int r = 0; r < 8; ++r) {
        float v = h[j][r] + bv;
        hidA[r + hiD][n] = (__bf16)(v > 0.f ? v : 0.f);
      }
    }
    __syncthreads();

    // ---------- GEMM4: logits = hidden @ Wc^T + bc (N padded 100 -> 112) ----------
    if (w < 7) {                       // wave-uniform branch: EXEC stays full
      v8f lg = zero8();
      int n = w * 16 + nrow;
      bool valid = (n < CLSv);
#pragma unroll
      for (int ks = 0; ks < NHv / 32; ++ks) {
        v16bf a = ldsFragA(&hidA[nrow][ks * 32 + hiA]);
        v16bf b = valid ? glbFragB(Wc + (size_t)n * NHv + ks * 32 + hiB) : zeroFragB();
        lg = WMMA_BF16(lg, a, b);
      }
      float bcv = valid ? bc[n] : 0.f;
#pragma unroll
      for (int r = 0; r < 8; ++r) logitsS[r + hiD][n] = lg[r] + bcv;
    }
    __syncthreads();

    // ---------- per-row log-softmax, loss, accuracy ----------
    if (tid < 16) {
      int r  = tid;
      int yb = y[m0 + r];
      float mx = logitsS[r][0];
      int   am = 0;
      for (int j = 1; j < CLSv; ++j) {
        float v = logitsS[r][j];
        if (v > mx) { mx = v; am = j; }
      }
      float se = 0.f;
      for (int j = 0; j < CLSv; ++j) se += __expf(logitsS[r][j] - mx);
      float logp = logitsS[r][yb] - mx - __logf(se);
      lossAcc += -logp;
      if (am == yb) atomicAdd(&out[1 + t], 1.0f / (float)BB);
    }
    // next iteration's leading __syncthreads() protects logitsS / lrI reuse
  }

  if (tid < 16) atomicAdd(&out[0], lossAcc * (1.0f / (float)(TT * BB)));
}

extern "C" void kernel_launch(void* const* d_in, const int* in_sizes, int n_in,
                              void* d_out, int out_size, void* d_ws, size_t ws_size,
                              hipStream_t stream) {
  (void)in_sizes; (void)n_in; (void)out_size;
  const float* x    = (const float*)d_in[0];
  const int*   y    = (const int*)  d_in[1];
  const int*   locs = (const int*)  d_in[2];
  const float* Wg   = (const float*)d_in[3];
  const float* bg   = (const float*)d_in[4];
  const float* Wl   = (const float*)d_in[5];
  const float* bl   = (const float*)d_in[6];
  const float* Wf   = (const float*)d_in[7];
  const float* bf   = (const float*)d_in[8];
  const float* Wxh  = (const float*)d_in[9];
  const float* Whh  = (const float*)d_in[10];
  const float* brnn = (const float*)d_in[11];
  const float* Wc   = (const float*)d_in[12];
  const float* bc   = (const float*)d_in[13];
  float* out = (float*)d_out;

  zero_out_k<<<1, 32, 0, stream>>>(out);

  if (ws_size >= (size_t)W_TOTAL * sizeof(__bf16)) {
    // Pre-convert all weights to bf16 in workspace (re-done every call:
    // deterministic, inputs are never mutated by the harness).
    __bf16* wsb = (__bf16*)d_ws;
    struct { const float* src; int ofs; int n; } cv[5] = {
      { Wg,  OFS_WG,  NFv * K1  },
      { Wf,  OFS_WF,  NFv * NFv },
      { Wxh, OFS_WXH, NHv * NFv },
      { Whh, OFS_WHH, NHv * NHv },
      { Wc,  OFS_WC,  CLSv * NHv },
    };
    for (int i = 0; i < 5; ++i) {
      int n4 = cv[i].n / 4;
      cvt_bf16_k<<<(n4 + 255) / 256, 256, 0, stream>>>(cv[i].src, wsb + cv[i].ofs, n4);
    }
    ram_fused<__bf16><<<BB / 16, 256, 0, stream>>>(
        x, y, locs,
        wsb + OFS_WG,  bg, Wl, bl,
        wsb + OFS_WF,  bf,
        wsb + OFS_WXH, wsb + OFS_WHH, brnn,
        wsb + OFS_WC,  bc, out);
  } else {
    // Fallback: stream f32 weights, convert in-register.
    ram_fused<float><<<BB / 16, 256, 0, stream>>>(
        x, y, locs, Wg, bg, Wl, bl, Wf, bf, Wxh, Whh, brnn, Wc, bc, out);
  }
}
